// NAMCell_72765335929659
// MI455X (gfx1250) — compile-verified, matrix operand order checked by reading, not run on hardware
//
#include <hip/hip_runtime.h>

#define DEV __device__ __forceinline__

typedef __bf16 bf16_t;
typedef __attribute__((ext_vector_type(16))) __bf16 v16bf;
typedef __attribute__((ext_vector_type(8)))  __bf16 v8bf;
typedef __attribute__((ext_vector_type(8)))  float  v8f;

static constexpr int B_ = 128, L_ = 128, H_ = 768, HEADS_ = 12, D_ = 64;
static constexpr int FFN_ = 3072, M_ = 128, R_ = 2;
static constexpr int BL_ = B_ * L_;        // 16384
static constexpr int BH_ = B_ * HEADS_;    // 1536

DEV bf16_t f2bf(float x) {
  unsigned u = __builtin_bit_cast(unsigned, x);
  u += 0x7FFFu + ((u >> 16) & 1u);
  unsigned short h = (unsigned short)(u >> 16);
  return __builtin_bit_cast(bf16_t, h);
}
DEV float sigmoidf_(float x) { return 1.f / (1.f + expf(-x)); }
DEV float softplusf_(float x) { return x > 20.f ? x : log1pf(expf(x)); }

DEV float bsum(float v, float* red, int n) {
  int t = threadIdx.x; red[t] = v; __syncthreads();
  for (int o = n >> 1; o > 0; o >>= 1) { if (t < o) red[t] += red[t + o]; __syncthreads(); }
  float r = red[0]; __syncthreads(); return r;
}
DEV float bmaxr(float v, float* red, int n) {
  int t = threadIdx.x; red[t] = v; __syncthreads();
  for (int o = n >> 1; o > 0; o >>= 1) { if (t < o) red[t] = fmaxf(red[t], red[t + o]); __syncthreads(); }
  float r = red[0]; __syncthreads(); return r;
}

DEV v16bf ldfrag(const bf16_t* p) {
  v8bf lo = *(const v8bf*)(p);
  v8bf hi = *(const v8bf*)(p + 16);
  return __builtin_shufflevector(lo, hi, 0,1,2,3,4,5,6,7,8,9,10,11,12,13,14,15);
}

// ---------------------------------------------------------------------------
// Generic bf16 WMMA GEMM:  C = act(alpha * A[M,K] @ Bt[N,K]^T + bias + resid)
// One wave computes a 32x64 tile (2 M-subtiles x 4 N-subtiles, 8 accumulators)
// REQUIRES: M % 32 == 0, N % 64 == 0, K % 32 == 0 (true for every call here),
// so the inner loops are branch-free and fully unrolled -> accumulators stay
// pinned in VGPRs and the loop body is pure loads + v_wmma.
// Fragment layout per ISA 7.12.2: lanes 0-15 hold K {0..7,16..23},
// lanes 16-31 hold K {8..15,24..31} of each 32-wide K step.
// ---------------------------------------------------------------------------
__global__ __launch_bounds__(32)
void gemm_wmma(const bf16_t* __restrict__ A, const bf16_t* __restrict__ Bt,
               const float* __restrict__ bias, const float* __restrict__ resid,
               float* __restrict__ Cf, bf16_t* __restrict__ Cbf,
               int Kdim, int lda, int ldb, int ldc,
               long long aBatch, long long bBatch, long long cBatch,
               float alpha, int act) {
  int lane = threadIdx.x;
  int lr = lane & 15;
  int hi = lane >> 4;
  int m0 = blockIdx.y * 32;
  int n0 = blockIdx.x * 64;
  long long zb = blockIdx.z;
  const bf16_t* Ab = A + zb * aBatch;
  const bf16_t* Bb = Bt + zb * bBatch;

  v8f acc[2][4];
#pragma unroll
  for (int mi = 0; mi < 2; mi++)
#pragma unroll
    for (int t = 0; t < 4; t++)
#pragma unroll
      for (int r = 0; r < 8; r++) acc[mi][t][r] = 0.f;

  const bf16_t* arow0 = Ab + (size_t)(m0 + lr) * lda + hi * 8;
  const bf16_t* arow1 = Ab + (size_t)(m0 + 16 + lr) * lda + hi * 8;
  const bf16_t* brow = Bb + (size_t)(n0 + lr) * ldb + hi * 8;

  for (int kb = 0; kb < Kdim; kb += 32) {
    v16bf af0 = ldfrag(arow0 + kb);
    v16bf af1 = ldfrag(arow1 + kb);
#pragma unroll
    for (int t = 0; t < 4; t++) {
      v16bf bfv = ldfrag(brow + (size_t)t * 16 * ldb + kb);
      acc[0][t] = __builtin_amdgcn_wmma_f32_16x16x32_bf16(
          false, af0, false, bfv, (short)0, acc[0][t], false, false);
      acc[1][t] = __builtin_amdgcn_wmma_f32_16x16x32_bf16(
          false, af1, false, bfv, (short)0, acc[1][t], false, false);
    }
  }

#pragma unroll
  for (int mi = 0; mi < 2; mi++) {
#pragma unroll
    for (int t = 0; t < 4; t++) {
      int col = n0 + t * 16 + lr;
      float bv = bias ? bias[col] : 0.f;
#pragma unroll
      for (int r = 0; r < 8; r++) {
        int row = m0 + mi * 16 + (hi ? r + 8 : r);
        size_t ci = (size_t)zb * cBatch + (size_t)row * ldc + col;
        float v = acc[mi][t][r] * alpha + bv;
        if (resid) v += resid[ci];
        if (act == 1) v = tanhf(v);
        else if (act == 2) v = sigmoidf_(v);
        else if (act == 3) v = v > 0.f ? v : 0.f;
        else if (act == 4) {
          float x = v;
          v = 0.5f * x * (1.f + tanhf(0.7978845608f * (x + 0.044715f * x * x * x)));
        }
        if (Cf)  Cf[ci] = v;
        if (Cbf) Cbf[ci] = f2bf(v);
      }
    }
  }
}

// --------------------------- data movement helpers -------------------------
__global__ void cast_bf_kernel(const float* __restrict__ x, bf16_t* __restrict__ y, int n) {
  int i = blockIdx.x * 256 + threadIdx.x;
  if (i < n) y[i] = f2bf(x[i]);
}
__global__ void transpose_cast_kernel(const float* __restrict__ w, bf16_t* __restrict__ wt,
                                      int K, int N) {
  int i = blockIdx.x * 256 + threadIdx.x;
  if (i < K * N) { int k = i / N, n = i % N; wt[(size_t)n * K + k] = f2bf(w[i]); }
}
__global__ void split_heads_kernel(const bf16_t* __restrict__ x, bf16_t* __restrict__ y) {
  int i = blockIdx.x * 256 + threadIdx.x;
  if (i >= BL_ * H_) return;
  int h = i % H_, bl = i / H_, b = bl / L_, l = bl % L_;
  int hd = h / D_, j = h % D_;
  y[((size_t)(b * HEADS_ + hd) * L_ + l) * D_ + j] = x[i];
}
__global__ void v_transpose_kernel(const bf16_t* __restrict__ x, bf16_t* __restrict__ y) {
  int i = blockIdx.x * 256 + threadIdx.x;
  if (i >= BL_ * H_) return;
  int h = i % H_, bl = i / H_, b = bl / L_, l = bl % L_;
  int hd = h / D_, j = h % D_;
  y[((size_t)(b * HEADS_ + hd) * D_ + j) * L_ + l] = x[i];
}
__global__ void merge_heads_kernel(const bf16_t* __restrict__ x, bf16_t* __restrict__ y) {
  int i = blockIdx.x * 256 + threadIdx.x;
  if (i >= BL_ * H_) return;
  int h = i % H_, bl = i / H_, b = bl / L_, l = bl % L_;
  int hd = h / D_, j = h % D_;
  y[i] = x[((size_t)(b * HEADS_ + hd) * L_ + l) * D_ + j];
}

// ------------------------------ layernorm ----------------------------------
__global__ __launch_bounds__(256)
void ln_kernel(const float* __restrict__ x, const float* __restrict__ g,
               const float* __restrict__ bb, bf16_t* __restrict__ ybf) {
  __shared__ float red[256];
  size_t row = blockIdx.x;
  const float* xr = x + row * H_;
  float s = 0.f;
  for (int i = threadIdx.x; i < H_; i += 256) s += xr[i];
  float mu = bsum(s, red, 256) / H_;
  float v = 0.f;
  for (int i = threadIdx.x; i < H_; i += 256) { float d = xr[i] - mu; v += d * d; }
  float var = bsum(v, red, 256) / H_;
  float rs = rsqrtf(var + 1e-6f);
  for (int i = threadIdx.x; i < H_; i += 256)
    ybf[row * H_ + i] = f2bf((xr[i] - mu) * rs * g[i] + bb[i]);
}

// ---------------------------- group attention ------------------------------
__global__ void ga_s_kernel(const float* __restrict__ q, const float* __restrict__ k,
                            const float* __restrict__ mask, float* __restrict__ logn) {
  int b = blockIdx.x, l = threadIdx.x;
  if (l >= L_ - 1) return;
  const float* qr = q + ((size_t)b * L_ + l) * H_;
  const float* kr = k + ((size_t)b * L_ + l + 1) * H_;
  float s = 0.f;
  for (int i = 0; i < H_; i++) s += qr[i] * kr[i];
  s *= 0.03608439182435161f;  // 1/sqrt(768)
  float nz = sigmoidf_(s) * mask[b * L_ + l] * mask[b * L_ + l + 1];
  logn[b * (L_ - 1) + l] = logf(nz + 1e-9f);
}
__global__ void cumsum_kernel(const float* __restrict__ logn, float* __restrict__ c) {
  int b = blockIdx.x;
  if (threadIdx.x) return;
  float a = 0.f; c[b * L_] = 0.f;
  for (int i = 1; i < L_; i++) { a += logn[b * (L_ - 1) + i - 1]; c[b * L_ + i] = a; }
}
__global__ void gp_kernel(const float* __restrict__ c, float* __restrict__ gp) {
  int bi = blockIdx.x; int b = bi / L_, i = bi % L_; int j = threadIdx.x;
  gp[(size_t)bi * L_ + j] = expf(-fabsf(c[b * L_ + i] - c[b * L_ + j]));
}

// ------------------------- masked softmax * group --------------------------
__global__ __launch_bounds__(128)
void softmax_p_kernel(const float* __restrict__ sc, const float* __restrict__ mask,
                      const float* __restrict__ gp, bf16_t* __restrict__ pbf) {
  __shared__ float red[128];
  int bhi = blockIdx.x;            // (b*HEADS+h)*L + i
  int i = bhi % L_;
  int bh = bhi / L_;
  int b = bh / HEADS_;
  int j = threadIdx.x;
  float v = sc[(size_t)bhi * L_ + j] + (1.f - mask[b * L_ + j]) * (-1.0e9f);
  float mx = bmaxr(v, red, 128);
  float e = expf(v - mx);
  float s = bsum(e, red, 128);
  float p = (e / s) * gp[((size_t)b * L_ + i) * L_ + j];
  float s2 = bsum(p, red, 128);
  pbf[(size_t)bhi * L_ + j] = f2bf(p / (s2 + 1e-9f));
}

// ------------------------------ reduce head --------------------------------
__global__ __launch_bounds__(256)
void redscore_kernel(const float* __restrict__ hid, const float* __restrict__ rw,
                     const float* __restrict__ rb, const float* __restrict__ mask,
                     float* __restrict__ out) {
  __shared__ float red[256];
  int row = blockIdx.x;
  float a = 0.f;
  for (int i = threadIdx.x; i < H_; i += 256) a += hid[(size_t)row * H_ + i] * rw[i];
  float s = bsum(a, red, 256);
  if (threadIdx.x == 0) out[row] = s + rb[0] + (1.f - mask[row]) * (-1.0e9f);
}
__global__ __launch_bounds__(128)
void softmax_rw_kernel(const float* __restrict__ sc, float* __restrict__ rw,
                       int* __restrict__ pos) {
  __shared__ float red[128];
  int b = blockIdx.x, t = threadIdx.x;
  float v = sc[b * L_ + t];
  float mx = bmaxr(v, red, 128);
  float e = expf(v - mx);
  float s = bsum(e, red, 128);
  rw[b * L_ + t] = e / s;
  if (t == 0) {
    float best = -3.4e38f; int bi = 0;
    for (int l = 0; l < L_; l++) { float x = sc[b * L_ + l]; if (x > best) { best = x; bi = l; } }
    pos[b] = bi;
  }
}
__global__ __launch_bounds__(128)
void context_kernel(const float* __restrict__ rw, const float* __restrict__ hid,
                    float* __restrict__ ctx, bf16_t* __restrict__ ctxbf) {
  int b = blockIdx.x; int h = blockIdx.y * 128 + threadIdx.x;
  float a = 0.f;
  for (int l = 0; l < L_; l++) a += rw[b * L_ + l] * hid[((size_t)b * L_ + l) * H_ + h];
  ctx[(size_t)b * H_ + h] = a;
  ctxbf[(size_t)b * H_ + h] = f2bf(a);
}
__global__ void matvec_small_kernel(const float* __restrict__ x, const float* __restrict__ w,
                                    const float* __restrict__ bias, float* __restrict__ out,
                                    int K, int N) {
  int b = blockIdx.x, t = threadIdx.x;
  if (t < N) {
    float a = bias ? bias[t] : 0.f;
    for (int k = 0; k < K; k++) a += x[(size_t)b * K + k] * w[(size_t)k * N + t];
    out[(size_t)b * N + t] = a;
  }
}

// ---------------------------- NTM addressing -------------------------------
__global__ __launch_bounds__(256)
void rownorm_kernel(const float* __restrict__ x, float* __restrict__ inv, int K) {
  __shared__ float red[256];
  size_t r = blockIdx.x;
  float a = 0.f;
  for (int i = threadIdx.x; i < K; i += 256) { float v = x[r * K + i]; a += v * v; }
  float s = bsum(a, red, 256);
  if (threadIdx.x == 0) inv[r] = 1.f / (sqrtf(s) + 1e-7f);
}
__global__ __launch_bounds__(256)
void sim_kernel(const float* __restrict__ key, const float* __restrict__ mem,
                const float* __restrict__ kinv, const float* __restrict__ minv,
                float* __restrict__ sim) {
  __shared__ float red[256];
  int bm = blockIdx.x; int b = bm / M_;
  float a = 0.f;
  for (int i = threadIdx.x; i < H_; i += 256)
    a += key[(size_t)b * H_ + i] * mem[(size_t)bm * H_ + i];
  float s = bsum(a, red, 256);
  if (threadIdx.x == 0) sim[bm] = s * kinv[b] * minv[bm];
}
__global__ __launch_bounds__(128)
void wc_kernel(const float* __restrict__ p6, const float* __restrict__ sim,
               float* __restrict__ wc) {
  __shared__ float red[128];
  int b = blockIdx.x, t = threadIdx.x;
  float beta = softplusf_(p6[b * 6 + 0]);
  float v = beta * sim[b * M_ + t];
  float mx = bmaxr(v, red, 128);
  float e = expf(v - mx);
  float s = bsum(e, red, 128);
  wc[b * M_ + t] = e / s;
}
__global__ __launch_bounds__(128)
void combine_kernel(const float* __restrict__ p6, const float* __restrict__ wc,
                    const float* __restrict__ wprev, float* __restrict__ wout) {
  __shared__ float wg[128];
  __shared__ float red[128];
  int b = blockIdx.x, t = threadIdx.x;
  float gate = sigmoidf_(p6[b * 6 + 1]);
  float gamma = 1.f + softplusf_(p6[b * 6 + 2]);
  float s3 = p6[b * 6 + 3], s4 = p6[b * 6 + 4], s5 = p6[b * 6 + 5];
  float mx = fmaxf(s3, fmaxf(s4, s5));
  float e3 = expf(s3 - mx), e4 = expf(s4 - mx), e5 = expf(s5 - mx);
  float ss = e3 + e4 + e5;
  float sh0 = e3 / ss, sh1 = e4 / ss, sh2 = e5 / ss;   // shifts -1, 0, +1
  wg[t] = gate * wc[b * M_ + t] + (1.f - gate) * wprev[b * M_ + t];
  __syncthreads();
  float wsh = sh0 * wg[(t + 1) & (M_ - 1)] + sh1 * wg[t] + sh2 * wg[(t + M_ - 1) & (M_ - 1)];
  float w = powf(wsh + 1e-7f, gamma);
  float sum = bsum(w, red, 128);
  wout[b * M_ + t] = w / (sum + 1e-7f);
}
__global__ __launch_bounds__(256)
void reads_kernel(const float* __restrict__ wr, const float* __restrict__ mem,
                  float* __restrict__ rf) {
  int b = blockIdx.x; int col = blockIdx.y * 256 + threadIdx.x;   // 0..R*H-1
  int r = col / H_, h = col % H_;
  float a = 0.f;
  for (int m = 0; m < M_; m++)
    a += wr[((size_t)r * B_ + b) * M_ + m] * mem[((size_t)b * M_ + m) * H_ + h];
  rf[(size_t)b * (R_ * H_) + col] = a;
}

// ------------------------------ packing ------------------------------------
__global__ __launch_bounds__(256)
void pack1_kernel(const float* __restrict__ ctx, const float* __restrict__ rf,
                  bf16_t* __restrict__ cat) {
  int b = blockIdx.x;
  for (int j = threadIdx.x; j < H_ + R_ * H_; j += 256) {
    float v = j < H_ ? ctx[(size_t)b * H_ + j] : rf[(size_t)b * (R_ * H_) + (j - H_)];
    cat[(size_t)b * (H_ + R_ * H_) + j] = f2bf(v);
  }
}
__global__ __launch_bounds__(256)
void pack2_kernel(const float* __restrict__ ctx, const float* __restrict__ re,
                  const float* __restrict__ ve, const float* __restrict__ rf,
                  bf16_t* __restrict__ cat) {
  int b = blockIdx.x;
  const int T = H_ * 3 + R_ * H_;   // 3840
  for (int j = threadIdx.x; j < T; j += 256) {
    float v;
    if (j < H_) v = ctx[(size_t)b * H_ + j];
    else if (j < 2 * H_) v = re[(size_t)b * H_ + (j - H_)];
    else if (j < 3 * H_) v = ve[(size_t)b * H_ + (j - 2 * H_)];
    else v = rf[(size_t)b * (R_ * H_) + (j - 3 * H_)];
    cat[(size_t)b * T + j] = f2bf(v);
  }
}
__global__ __launch_bounds__(256)
void resval_kernel(const float* __restrict__ result, const float* __restrict__ valid,
                   const float* __restrict__ res_w, const float* __restrict__ res_b,
                   const float* __restrict__ val_w, const float* __restrict__ val_b,
                   float* __restrict__ resenc, float* __restrict__ valenc) {
  int b = blockIdx.x;
  float r = result[b], v = valid[b];
  for (int h = threadIdx.x; h < H_; h += 256) {
    resenc[(size_t)b * H_ + h] = r * res_w[h] + v * res_w[H_ + h] + res_b[h];
    valenc[(size_t)b * H_ + h] = v * val_w[h] + val_b[h];
  }
}

// ------------------------- digit ALU + op mixing ---------------------------
__global__ void assemble_kernel(const int* __restrict__ tok, const float* __restrict__ mask,
                                const int* __restrict__ pos, const float* __restrict__ opl,
                                float* __restrict__ o_result, float* __restrict__ o_valid,
                                float* __restrict__ o_left, float* __restrict__ o_right) {
  int b = blockIdx.x;
  if (threadIdx.x != 0) return;
  int P = pos[b];
  float vals[2];
  for (int side = 0; side < 2; side++) {
    float total = 0.f;
    for (int l = 0; l < L_; l++) {
      int id = tok[b * L_ + l];
      float isd = (id >= 4 && id <= 13) ? 1.f : 0.f;
      float pm = side == 0 ? (l < P ? 1.f : 0.f) : (l > P ? 1.f : 0.f);
      total += isd * pm * mask[b * L_ + l];
    }
    float cum = 0.f, val = 0.f;
    for (int l = 0; l < L_; l++) {
      int id = tok[b * L_ + l];
      float isd = (id >= 4 && id <= 13) ? 1.f : 0.f;
      float pm = side == 0 ? (l < P ? 1.f : 0.f) : (l > P ? 1.f : 0.f);
      float dm = isd * pm * mask[b * L_ + l];
      cum += dm;
      float dval = (float)(id - 4) * isd;
      val += dval * powf(10.f, (total - cum) * dm) * dm;
    }
    vals[side] = val;
  }
  float lv = vals[0], rv = vals[1];
  float addr = lv + rv, subr = lv - rv, mulr = lv * rv;
  float ab = fabsf(rv);
  float dv = ab > 1e-7f ? 1.f : 0.f;
  float divr = dv > 0.f ? lv / rv : 0.f;
  float o0 = opl[b * 4 + 0], o1 = opl[b * 4 + 1], o2 = opl[b * 4 + 2], o3 = opl[b * 4 + 3];
  float mx = fmaxf(fmaxf(o0, o1), fmaxf(o2, o3));
  float e0 = expf(o0 - mx), e1 = expf(o1 - mx), e2 = expf(o2 - mx), e3 = expf(o3 - mx);
  float es = e0 + e1 + e2 + e3;
  float p0 = e0 / es, p1 = e1 / es, p2 = e2 / es, p3 = e3 / es;
  o_result[b] = p0 * addr + p1 * subr + p2 * mulr + p3 * divr;
  o_valid[b] = p0 + p1 + p2 + p3 * dv;
  o_left[b] = lv; o_right[b] = rv;
}

// ---------------------------- memory update --------------------------------
__global__ __launch_bounds__(256)
void memupdate_kernel(const float* __restrict__ mem, const float* __restrict__ ww,
                      const float* __restrict__ er, const float* __restrict__ av,
                      float* __restrict__ nm) {
  int bm = blockIdx.x; int b = bm / M_;
  float w = ww[bm];
  for (int h = threadIdx.x; h < H_; h += 256) {
    float e = er[(size_t)b * H_ + h], a = av[(size_t)b * H_ + h];
    nm[(size_t)bm * H_ + h] = mem[(size_t)bm * H_ + h] * (1.f - w * e) + w * a;
  }
}

// ===========================================================================
extern "C" void kernel_launch(void* const* d_in, const int* in_sizes, int n_in,
                              void* d_out, int out_size, void* d_ws, size_t ws_size,
                              hipStream_t stream) {
  (void)in_sizes; (void)n_in; (void)out_size; (void)ws_size;

  const float* in_hidden = (const float*)d_in[0];
  const float* in_mask   = (const float*)d_in[1];
  const int*   in_tok    = (const int*)d_in[2];
  const float* in_mem    = (const float*)d_in[3];
  const float* in_readw  = (const float*)d_in[4];
  const float* in_writew = (const float*)d_in[5];
  const float* ga_wq = (const float*)d_in[6];
  const float* ga_wk = (const float*)d_in[7];
  const float* mha_wq = (const float*)d_in[8];
  const float* mha_bq = (const float*)d_in[9];
  const float* mha_wk = (const float*)d_in[10];
  const float* mha_bk = (const float*)d_in[11];
  const float* mha_wv = (const float*)d_in[12];
  const float* mha_bv = (const float*)d_in[13];
  const float* mha_wo = (const float*)d_in[14];
  const float* mha_bo = (const float*)d_in[15];
  const float* ln1_g = (const float*)d_in[16];
  const float* ln1_b = (const float*)d_in[17];
  const float* ffn_w1 = (const float*)d_in[18];
  const float* ffn_b1 = (const float*)d_in[19];
  const float* ffn_w2 = (const float*)d_in[20];
  const float* ffn_b2 = (const float*)d_in[21];
  const float* ln2_g = (const float*)d_in[22];
  const float* ln2_b = (const float*)d_in[23];
  const float* red_w = (const float*)d_in[24];
  const float* red_b = (const float*)d_in[25];
  const float* op_w = (const float*)d_in[26];
  const float* op_b = (const float*)d_in[27];
  const float* rh_key_w = (const float*)d_in[28];
  const float* rh_key_b = (const float*)d_in[29];
  const float* rh_addr_w = (const float*)d_in[30];
  const float* rh_addr_b = (const float*)d_in[31];
  const float* wh_key_w = (const float*)d_in[32];
  const float* wh_key_b = (const float*)d_in[33];
  const float* wh_addr_w = (const float*)d_in[34];
  const float* wh_addr_b = (const float*)d_in[35];
  const float* wh_erase_w = (const float*)d_in[36];
  const float* wh_erase_b = (const float*)d_in[37];
  const float* wh_add_w = (const float*)d_in[38];
  const float* wh_add_b = (const float*)d_in[39];
  const float* ctrl_w = (const float*)d_in[40];
  const float* ctrl_b = (const float*)d_in[41];
  const float* res_w = (const float*)d_in[42];
  const float* res_b = (const float*)d_in[43];
  const float* val_w = (const float*)d_in[44];
  const float* val_b = (const float*)d_in[45];
  const float* su_w = (const float*)d_in[46];
  const float* su_b = (const float*)d_in[47];
  const float* halt_w = (const float*)d_in[48];
  const float* halt_b = (const float*)d_in[49];

  // ---- output layout
  float* out = (float*)d_out;
  float* o_hidden = out;
  float* o_result = out + (size_t)BL_ * H_;
  float* o_valid  = o_result + B_;
  float* o_opl    = o_valid + B_;
  float* o_q      = o_opl + 4 * B_;
  float* o_left   = o_q + 2 * B_;
  float* o_right  = o_left + B_;
  float* o_newmem = o_right + B_;

  // ---- workspace allocator (with explicit region reuse)
  char* W = (char*)d_ws;
  size_t off = 0;
  auto alloc = [&](size_t bytes) -> char* {
    char* p = W + off;
    off += (bytes + 255) & ~(size_t)255;
    return p;
  };
  const size_t BLH = (size_t)BL_ * H_;

  bf16_t* hidbf = (bf16_t*)alloc(BLH * 2);            // reused later: merged attn bf16
  bf16_t* gaqT = (bf16_t*)alloc((size_t)H_ * H_ * 2);
  bf16_t* gakT = (bf16_t*)alloc((size_t)H_ * H_ * 2);
  bf16_t* wqT  = (bf16_t*)alloc((size_t)H_ * H_ * 2);
  bf16_t* wkT  = (bf16_t*)alloc((size_t)H_ * H_ * 2);
  bf16_t* wvT  = (bf16_t*)alloc((size_t)H_ * H_ * 2);
  bf16_t* woT  = (bf16_t*)alloc((size_t)H_ * H_ * 2);
  bf16_t* f1T  = (bf16_t*)alloc((size_t)H_ * FFN_ * 2);
  bf16_t* f2T  = (bf16_t*)alloc((size_t)H_ * FFN_ * 2);
  bf16_t* rhkT0 = (bf16_t*)alloc((size_t)H_ * H_ * 2);
  bf16_t* rhkT1 = (bf16_t*)alloc((size_t)H_ * H_ * 2);
  bf16_t* whkT = (bf16_t*)alloc((size_t)H_ * H_ * 2);
  bf16_t* wheT = (bf16_t*)alloc((size_t)H_ * H_ * 2);
  bf16_t* whaT = (bf16_t*)alloc((size_t)H_ * H_ * 2);
  bf16_t* ctrlT = (bf16_t*)alloc((size_t)(H_ + R_ * H_) * H_ * 2);
  bf16_t* suT  = (bf16_t*)alloc((size_t)(3 * H_ + R_ * H_) * H_ * 2);
  bf16_t* ln1bf = (bf16_t*)alloc(BLH * 2);            // reused later: attnout bf16
  char* big1 = alloc((size_t)BH_ * L_ * L_ * 4);      // qga+kga | scores | ffn1bf
  float* qga = (float*)big1;
  float* kga = (float*)(big1 + BLH * 4);
  float* scoresF = (float*)big1;
  bf16_t* ffn1bf = (bf16_t*)big1;
  char* qkv = alloc(3 * BLH * 2);                     // qbf,kbf,vbf | pbf
  bf16_t* qbf = (bf16_t*)qkv;
  bf16_t* kbf = (bf16_t*)(qkv + BLH * 2);
  bf16_t* vbf = (bf16_t*)(qkv + 2 * BLH * 2);
  bf16_t* pbf = (bf16_t*)qkv;
  char* hds = alloc(3 * BLH * 2);
  bf16_t* qh = (bf16_t*)hds;
  bf16_t* kh = (bf16_t*)(hds + BLH * 2);
  bf16_t* vT = (bf16_t*)(hds + 2 * BLH * 2);
  float* hidden1 = (float*)alloc(BLH * 4);
  bf16_t* ln2bf = (bf16_t*)alloc(BLH * 2);
  float* gpbuf = (float*)alloc((size_t)B_ * L_ * L_ * 4);
  float* logn = (float*)alloc((size_t)B_ * (L_ - 1) * 4);
  float* cvec = (float*)alloc((size_t)B_ * L_ * 4);
  float* redsc = (float*)alloc((size_t)BL_ * 4);
  float* rwbuf = (float*)alloc((size_t)BL_ * 4);
  int* posbuf = (int*)alloc((size_t)B_ * 4);
  float* ctx = (float*)alloc((size_t)B_ * H_ * 4);
  bf16_t* ctxbf = (bf16_t*)alloc((size_t)B_ * H_ * 2);
  float* keybuf = (float*)alloc((size_t)B_ * H_ * 4);
  float* p6 = (float*)alloc((size_t)B_ * 6 * 4);
  float* kinv = (float*)alloc((size_t)B_ * 4);
  float* minv = (float*)alloc((size_t)B_ * M_ * 4);
  float* simb = (float*)alloc((size_t)B_ * M_ * 4);
  float* wcb = (float*)alloc((size_t)B_ * M_ * 4);
  float* wread = (float*)alloc((size_t)R_ * B_ * M_ * 4);
  float* wwrite = (float*)alloc((size_t)B_ * M_ * 4);
  float* readsf = (float*)alloc((size_t)B_ * R_ * H_ * 4);
  bf16_t* cat1 = (bf16_t*)alloc((size_t)B_ * (H_ + R_ * H_) * 2);
  float* ctrl = (float*)alloc((size_t)B_ * H_ * 4);
  bf16_t* ctrlbf = (bf16_t*)alloc((size_t)B_ * H_ * 2);
  float* eraseb = (float*)alloc((size_t)B_ * H_ * 4);
  float* addvb = (float*)alloc((size_t)B_ * H_ * 4);
  float* resenc = (float*)alloc((size_t)B_ * H_ * 4);
  float* valenc = (float*)alloc((size_t)B_ * H_ * 4);
  bf16_t* cat2 = (bf16_t*)alloc((size_t)B_ * (3 * H_ + R_ * H_) * 2);
  float* stateb = (float*)alloc((size_t)B_ * H_ * 4);
  bf16_t* attnbf = ln1bf;   // region reuse
  bf16_t* attn2bf = hidbf;  // region reuse

  auto gemm = [&](const bf16_t* A, const bf16_t* Bt, const float* bias, const float* resid,
                  float* Cf, bf16_t* Cbf, int M, int N, int K, int lda, int ldb, int ldc,
                  long long aB, long long bB, long long cB, int batch, float alpha, int act) {
    dim3 g(N / 64, M / 32, batch);
    gemm_wmma<<<g, 32, 0, stream>>>(A, Bt, bias, resid, Cf, Cbf, K,
                                    lda, ldb, ldc, aB, bB, cB, alpha, act);
  };
  auto tpose = [&](const float* w, bf16_t* wt, int K, int N) {
    int n = K * N;
    transpose_cast_kernel<<<(n + 255) / 256, 256, 0, stream>>>(w, wt, K, N);
  };

  // ---- 0: casts + weight transposes (bf16, [N,K] for WMMA B-fragments)
  cast_bf_kernel<<<(int)((BLH + 255) / 256), 256, 0, stream>>>(in_hidden, hidbf, (int)BLH);
  tpose(ga_wq, gaqT, H_, H_);  tpose(ga_wk, gakT, H_, H_);
  tpose(mha_wq, wqT, H_, H_);  tpose(mha_wk, wkT, H_, H_);
  tpose(mha_wv, wvT, H_, H_);  tpose(mha_wo, woT, H_, H_);
  tpose(ffn_w1, f1T, H_, FFN_); tpose(ffn_w2, f2T, FFN_, H_);
  tpose(rh_key_w, rhkT0, H_, H_);
  tpose(rh_key_w + (size_t)H_ * H_, rhkT1, H_, H_);
  tpose(wh_key_w, whkT, H_, H_); tpose(wh_erase_w, wheT, H_, H_); tpose(wh_add_w, whaT, H_, H_);
  tpose(ctrl_w, ctrlT, H_ + R_ * H_, H_);
  tpose(su_w, suT, 3 * H_ + R_ * H_, H_);

  // ---- 1: group attention
  gemm(hidbf, gaqT, nullptr, nullptr, qga, nullptr, BL_, H_, H_, H_, H_, H_, 0, 0, 0, 1, 1.f, 0);
  gemm(hidbf, gakT, nullptr, nullptr, kga, nullptr, BL_, H_, H_, H_, H_, H_, 0, 0, 0, 1, 1.f, 0);
  ga_s_kernel<<<B_, 128, 0, stream>>>(qga, kga, in_mask, logn);
  cumsum_kernel<<<B_, 32, 0, stream>>>(logn, cvec);
  gp_kernel<<<B_ * L_, 128, 0, stream>>>(cvec, gpbuf);

  // ---- 2: ln1 + q/k/v projections
  ln_kernel<<<BL_, 256, 0, stream>>>(in_hidden, ln1_g, ln1_b, ln1bf);
  gemm(ln1bf, wqT, mha_bq, nullptr, nullptr, qbf, BL_, H_, H_, H_, H_, H_, 0, 0, 0, 1, 1.f, 0);
  gemm(ln1bf, wkT, mha_bk, nullptr, nullptr, kbf, BL_, H_, H_, H_, H_, H_, 0, 0, 0, 1, 1.f, 0);
  gemm(ln1bf, wvT, mha_bv, nullptr, nullptr, vbf, BL_, H_, H_, H_, H_, H_, 0, 0, 0, 1, 1.f, 0);
  int nBLH = (int)((BLH + 255) / 256);
  split_heads_kernel<<<nBLH, 256, 0, stream>>>(qbf, qh);
  split_heads_kernel<<<nBLH, 256, 0, stream>>>(kbf, kh);
  v_transpose_kernel<<<nBLH, 256, 0, stream>>>(vbf, vT);

  // ---- 3: attention (batched WMMA), softmax*group, p@v, out proj, FFN
  gemm(qh, kh, nullptr, nullptr, scoresF, nullptr, L_, L_, D_, D_, D_, L_,
       (long long)L_ * D_, (long long)L_ * D_, (long long)L_ * L_, BH_, 0.125f, 0);
  softmax_p_kernel<<<BH_ * L_, 128, 0, stream>>>(scoresF, in_mask, gpbuf, pbf);
  gemm(pbf, vT, nullptr, nullptr, nullptr, attnbf, L_, D_, L_, L_, L_, D_,
       (long long)L_ * L_, (long long)D_ * L_, (long long)L_ * D_, BH_, 1.f, 0);
  merge_heads_kernel<<<nBLH, 256, 0, stream>>>(attnbf, attn2bf);
  gemm(attn2bf, woT, mha_bo, in_hidden, hidden1, nullptr, BL_, H_, H_, H_, H_, H_,
       0, 0, 0, 1, 1.f, 0);
  ln_kernel<<<BL_, 256, 0, stream>>>(hidden1, ln2_g, ln2_b, ln2bf);
  gemm(ln2bf, f1T, ffn_b1, nullptr, nullptr, ffn1bf, BL_, FFN_, H_, H_, H_, FFN_,
       0, 0, 0, 1, 1.f, 4 /*gelu*/);
  gemm(ffn1bf, f2T, ffn_b2, hidden1, o_hidden, nullptr, BL_, H_, FFN_, FFN_, FFN_, H_,
       0, 0, 0, 1, 1.f, 0);

  // ---- 4: reduction head, context, op logits
  redscore_kernel<<<BL_, 256, 0, stream>>>(o_hidden, red_w, red_b, in_mask, redsc);
  softmax_rw_kernel<<<B_, 128, 0, stream>>>(redsc, rwbuf, posbuf);
  context_kernel<<<dim3(B_, H_ / 128), 128, 0, stream>>>(rwbuf, o_hidden, ctx, ctxbf);
  matvec_small_kernel<<<B_, 32, 0, stream>>>(ctx, op_w, op_b, o_opl, H_, 4);

  // ---- 5: read heads
  rownorm_kernel<<<B_ * M_, 256, 0, stream>>>(in_mem, minv, H_);
  for (int r = 0; r < R_; r++) {
    bf16_t* kT = r == 0 ? rhkT0 : rhkT1;
    gemm(ctxbf, kT, rh_key_b + (size_t)r * H_, nullptr, keybuf, nullptr,
         B_, H_, H_, H_, H_, H_, 0, 0, 0, 1, 1.f, 1 /*tanh*/);
    rownorm_kernel<<<B_, 256, 0, stream>>>(keybuf, kinv, H_);
    matvec_small_kernel<<<B_, 32, 0, stream>>>(ctx, rh_addr_w + (size_t)r * H_ * 6,
                                               rh_addr_b + (size_t)r * 6, p6, H_, 6);
    sim_kernel<<<B_ * M_, 256, 0, stream>>>(keybuf, in_mem, kinv, minv, simb);
    wc_kernel<<<B_, 128, 0, stream>>>(p6, simb, wcb);
    combine_kernel<<<B_, 128, 0, stream>>>(p6, wcb, in_readw + (size_t)r * B_ * M_,
                                           wread + (size_t)r * B_ * M_);
  }
  reads_kernel<<<dim3(B_, (R_ * H_) / 256), 256, 0, stream>>>(wread, in_mem, readsf);
  pack1_kernel<<<B_, 256, 0, stream>>>(ctx, readsf, cat1);
  gemm(cat1, ctrlT, ctrl_b, nullptr, ctrl, ctrlbf, B_, H_, H_ + R_ * H_,
       H_ + R_ * H_, H_ + R_ * H_, H_, 0, 0, 0, 1, 1.f, 3 /*relu*/);

  // ---- 6: digit ALU + op mixing
  assemble_kernel<<<B_, 32, 0, stream>>>(in_tok, in_mask, posbuf, o_opl,
                                         o_result, o_valid, o_left, o_right);

  // ---- 7: write head + memory update
  gemm(ctrlbf, whkT, wh_key_b, nullptr, keybuf, nullptr, B_, H_, H_, H_, H_, H_,
       0, 0, 0, 1, 1.f, 1 /*tanh*/);
  rownorm_kernel<<<B_, 256, 0, stream>>>(keybuf, kinv, H_);
  matvec_small_kernel<<<B_, 32, 0, stream>>>(ctrl, wh_addr_w, wh_addr_b, p6, H_, 6);
  sim_kernel<<<B_ * M_, 256, 0, stream>>>(keybuf, in_mem, kinv, minv, simb);
  wc_kernel<<<B_, 128, 0, stream>>>(p6, simb, wcb);
  combine_kernel<<<B_, 128, 0, stream>>>(p6, wcb, in_writew, wwrite);
  gemm(ctrlbf, wheT, wh_erase_b, nullptr, eraseb, nullptr, B_, H_, H_, H_, H_, H_,
       0, 0, 0, 1, 1.f, 2 /*sigmoid*/);
  gemm(ctrlbf, whaT, wh_add_b, nullptr, addvb, nullptr, B_, H_, H_, H_, H_, H_,
       0, 0, 0, 1, 1.f, 1 /*tanh*/);
  memupdate_kernel<<<B_ * M_, 256, 0, stream>>>(in_mem, wwrite, eraseb, addvb, o_newmem);

  // ---- 8: state + halt
  resval_kernel<<<B_, 256, 0, stream>>>(o_result, o_valid, res_w, res_b, val_w, val_b,
                                        resenc, valenc);
  pack2_kernel<<<B_, 256, 0, stream>>>(ctx, resenc, valenc, readsf, cat2);
  gemm(cat2, suT, su_b, nullptr, stateb, nullptr, B_, H_, 3 * H_ + R_ * H_,
       3 * H_ + R_ * H_, 3 * H_ + R_ * H_, H_, 0, 0, 0, 1, 1.f, 0);
  matvec_small_kernel<<<B_, 32, 0, stream>>>(stateb, halt_w, halt_b, o_q, H_, 2);
}
